// NodeClassifier_73624329388568
// MI455X (gfx1250) — compile-verified, matrix operand order checked by reading, not run on hardware
//
#include <hip/hip_runtime.h>

typedef __attribute__((ext_vector_type(16))) _Float16 v16h;
typedef __attribute__((ext_vector_type(8)))  float    v8f;

#define NNODES 100000
#define NEDGES 800000

#if defined(__has_builtin)
#  if __has_builtin(__builtin_amdgcn_global_load_async_to_lds_b128)
#    define USE_ASYNC_STAGE 1
#  endif
#endif

// exact parameter types of __builtin_amdgcn_global_load_async_to_lds_b128:
//   (int __vector(4) addrspace(1)*, int __vector(4) addrspace(3)*, Imm int, Imm int)
typedef int vint4 __attribute__((__vector_size__(16)));
typedef __attribute__((address_space(1))) vint4 g_vint4;
typedef __attribute__((address_space(3))) vint4 l_vint4;

__device__ __forceinline__ void atomAddF(float* p, float v) {
  // device-scope relaxed float add -> hardware global_atomic_add_f32 (no CAS loop)
  __hip_atomic_fetch_add(p, v, __ATOMIC_RELAXED, __HIP_MEMORY_SCOPE_AGENT);
}

__device__ __forceinline__ void wait_async0() {
#if defined(__has_builtin) && __has_builtin(__builtin_amdgcn_s_wait_asynccnt)
  __builtin_amdgcn_s_wait_asynccnt(0);
#else
  asm volatile("s_wait_asynccnt 0x0" ::: "memory");
#endif
}

// ---------------------------------------------------------------- utilities

__global__ void zero_kernel(float* __restrict__ p, long n) {
  long i = blockIdx.x * 256L + threadIdx.x;
  long stride = gridDim.x * 256L;
  for (; i < n; i += stride) p[i] = 0.0f;
}

__global__ void deg_kernel(const int* __restrict__ ei, float* __restrict__ deg, int e_total) {
  int e = blockIdx.x * 256 + threadIdx.x;
  if (e >= e_total) return;
  atomAddF(deg + ei[e_total + e], 1.0f);   // dst row of edge_index
}

__global__ void inv_kernel(float* __restrict__ deg, int n) {
  int i = blockIdx.x * 256 + threadIdx.x;
  if (i < n) deg[i] = 1.0f / fmaxf(deg[i], 1.0f);
}

// ------------------------------------------------- weight split into B-frags
// B[k][n] = (k<128 ? Wn[n][k] : Wr[n][k-128]); split into f16 hi/lo halves,
// stored in WMMA B-fragment order: halves[((half*64 + frag)*32 + lane)*16 + e]
__global__ void prep_w_kernel(const float* __restrict__ Wn0, const float* __restrict__ Wr0,
                              const float* __restrict__ Wn1, const float* __restrict__ Wr1,
                              const float* __restrict__ Wn2, const float* __restrict__ Wr2,
                              _Float16* __restrict__ wsW) {
  int idx = blockIdx.x * blockDim.x + threadIdx.x;   // 3 layers * 64 frags * 32 lanes
  if (idx >= 3 * 64 * 32) return;
  int lane = idx & 31;
  int frag = (idx >> 5) & 63;
  int l    = idx >> 11;
  int kt = frag >> 3, nt = frag & 7;
  const float* Wn = (l == 0) ? Wn0 : (l == 1) ? Wn1 : Wn2;
  const float* Wr = (l == 0) ? Wr0 : (l == 1) ? Wr1 : Wr2;
  int col = nt * 16 + (lane & 15);
  _Float16* hi = wsW + ((size_t)(l * 2 + 0) * 64 + frag) * 512 + lane * 16;
  _Float16* lo = wsW + ((size_t)(l * 2 + 1) * 64 + frag) * 512 + lane * 16;
  for (int e = 0; e < 16; ++e) {
    int klocal = (e < 8 ? e : e + 8) + ((lane >= 16) ? 8 : 0);
    int k = kt * 32 + klocal;
    float w = (k < 128) ? Wn[col * 128 + k] : Wr[col * 128 + (k - 128)];
    _Float16 h = (_Float16)w;
    hi[e] = h;
    lo[e] = (_Float16)(w - (float)h);
  }
}

// ------------------------------------------------------------ edge scatter
// one wave per edge: 32 lanes x float4 = 128 channels; L2-resident HW atomics
__global__ __launch_bounds__(256) void scatter_kernel(const float* __restrict__ h,
                                                      const int* __restrict__ ei,
                                                      float* __restrict__ agg, int e_total) {
  long gid = blockIdx.x * 256L + threadIdx.x;
  int e = (int)(gid >> 5);
  if (e >= e_total) return;
  int c = ((int)gid & 31) * 4;
  int src = ei[e];
  int dst = ei[e_total + e];
  const float4 v = *(const float4*)(h + (size_t)src * 128 + c);
  float* p = agg + (size_t)dst * 128 + c;
  atomAddF(p + 0, v.x);
  atomAddF(p + 1, v.y);
  atomAddF(p + 2, v.z);
  atomAddF(p + 3, v.w);
}

// --------------------------------------------------------------- WMMA GEMM
// out(64 x 128) = [mean | h](64 x 256) @ B(256 x 128), split-f16 with f32 acc.
// LDS swizzle in float2 units: pair p of row r stored at r*128 + (p ^ ((r&15)<<1))
__device__ __forceinline__ int sm_idx2(int row, int k) {
  return (row * 128 + ((k >> 1) ^ ((row & 15) << 1))) * 2;
}

__global__ __launch_bounds__(128) void gemm_kernel(const float* agg, const float* __restrict__ hin,
                                                   const float* __restrict__ inv,
                                                   const _Float16* __restrict__ wsW,
                                                   float* out, float* __restrict__ stats, int n) {
  __shared__ float smem[64 * 256];            // 64 KB: 64 nodes x 256 K (swizzled, raw)
  const int tid = threadIdx.x;
  const int blockbase = blockIdx.x * 64;

  // ---- stage RAW tile: cols 0..127 = agg, cols 128..255 = h (scaling applied later)
#ifdef USE_ASYNC_STAGE
  for (int idx = tid; idx < 64 * 64; idx += 128) {
    int row = idx >> 6;
    int c4 = (idx & 63) * 4;
    int gnode = blockbase + row;
    int gn = (gnode < n) ? gnode : (n - 1);              // clamp: no OOB reads
    const float* gp = (c4 < 128) ? (agg + (size_t)gn * 128 + c4)
                                 : (hin + (size_t)gn * 128 + (c4 - 128));
    const float* lp = smem + sm_idx2(row, c4);
    // AS1 pointer value == flat value; AS3 pointer value == low 32 bits of flat (LDS offset)
    __builtin_amdgcn_global_load_async_to_lds_b128(
        (g_vint4*)(uintptr_t)gp, (l_vint4*)(unsigned)(uintptr_t)lp, 0, 0);
  }
  wait_async0();
#else
  for (int idx = tid; idx < 64 * 64; idx += 128) {
    int row = idx >> 6;
    int c4 = (idx & 63) * 4;
    int gnode = blockbase + row;
    int gn = (gnode < n) ? gnode : (n - 1);
    const float4 v = (c4 < 128) ? *(const float4*)(agg + (size_t)gn * 128 + c4)
                                : *(const float4*)(hin + (size_t)gn * 128 + (c4 - 128));
    *(float4*)(smem + sm_idx2(row, c4)) = v;
  }
#endif
  __syncthreads();

  const int wid = tid >> 5, lane = tid & 31;
  const int hh = lane >> 4;          // half-wave select (K offset +8)
  const int mrel = lane & 15;
  const int arow = wid * 16 + mrel;  // A row within the 64-row tile

  // per-row scaling: mean part gets 1/deg, h part gets 1; padded rows -> 0
  float scm = 0.f, sch = 0.f;
  {
    int gnode = blockbase + arow;
    if (gnode < n) { scm = inv[gnode]; sch = 1.0f; }
  }

  v8f acc[8] = {};
  for (int kt = 0; kt < 8; ++kt) {
    const float sc = (kt < 4) ? scm : sch;
    // build A hi/lo fragments from LDS (scale + f32 -> f16 split)
    v16h ahi, alo;
#pragma unroll
    for (int j = 0; j < 8; ++j) {
      int k0 = (j < 4 ? 2 * j : 2 * j + 8) + hh * 8;
      const float2 v = *(const float2*)(smem + sm_idx2(arow, kt * 32 + k0));
      float vx = v.x * sc, vy = v.y * sc;
      _Float16 hx = (_Float16)vx;
      _Float16 hy = (_Float16)vy;
      ahi[2 * j] = hx;           ahi[2 * j + 1] = hy;
      alo[2 * j] = (_Float16)(vx - (float)hx);
      alo[2 * j + 1] = (_Float16)(vy - (float)hy);
    }
#pragma unroll
    for (int nt = 0; nt < 8; ++nt) {
      const int frag = kt * 8 + nt;
      const v16h bhi = *(const v16h*)(wsW + ((size_t)(0 * 64 + frag) * 32 + lane) * 16);
      const v16h blo = *(const v16h*)(wsW + ((size_t)(1 * 64 + frag) * 32 + lane) * 16);
      acc[nt] = __builtin_amdgcn_wmma_f32_16x16x32_f16(false, ahi, false, bhi, (short)0, acc[nt], false, false);
      acc[nt] = __builtin_amdgcn_wmma_f32_16x16x32_f16(false, alo, false, bhi, (short)0, acc[nt], false, false);
      acc[nt] = __builtin_amdgcn_wmma_f32_16x16x32_f16(false, ahi, false, blo, (short)0, acc[nt], false, false);
    }
  }

  // ---- store (in-place into agg) + BatchNorm partial sums
#pragma unroll
  for (int nt = 0; nt < 8; ++nt) {
    const int col = nt * 16 + mrel;
    float s = 0.f, sq = 0.f;
#pragma unroll
    for (int r = 0; r < 8; ++r) {
      float val = acc[nt][r];
      int gnode = blockbase + wid * 16 + hh * 8 + r;   // C layout: M = r (+8 for hi half)
      if (gnode < n) out[(size_t)gnode * 128 + col] = val;
      s += val;            // padded rows contribute exact zeros (sc == 0)
      sq += val * val;
    }
    atomAddF(stats + col, s);
    atomAddF(stats + 128 + col, sq);
  }
}

// ----------------------------------------------------------- BN finalize/apply
__global__ void bn_finalize_kernel(const float* __restrict__ stats, const float* __restrict__ g,
                                   const float* __restrict__ be, float* __restrict__ ss, float invN) {
  int c = threadIdx.x;             // 128 threads
  float m = stats[c] * invN;
  float var = stats[128 + c] * invN - m * m;
  float sc = g[c] * rsqrtf(var + 1e-5f);
  ss[c] = sc;
  ss[128 + c] = be[c] - m * sc;
}

__global__ void bn_relu_kernel(const float* __restrict__ x, const float* __restrict__ ss,
                               float* __restrict__ y, long total) {
  long idx = blockIdx.x * 256L + threadIdx.x;
  if (idx >= total) return;
  int c = (int)(idx & 127);
  y[idx] = fmaxf(fmaf(x[idx], ss[c], ss[128 + c]), 0.f);
}

// -------------------------------------------------------------- MLP head
__global__ __launch_bounds__(256) void head_kernel(const float* __restrict__ h,
                                                   const float* __restrict__ W1,
                                                   const float* __restrict__ b1,
                                                   const float* __restrict__ W2,
                                                   const float* __restrict__ b2,
                                                   float* __restrict__ outp, int n) {
  __shared__ float w1s[64 * 129];   // padded stride: conflict-free column reads
  __shared__ float b1s[64];
  __shared__ float w2s[2 * 64];
  __shared__ float b2s[2];
  __shared__ float rowbuf[8][128];
  __shared__ float hidbuf[8][64];
  const int tid = threadIdx.x;
  for (int i = tid; i < 64 * 128; i += 256) {
    int j = i >> 7, c = i & 127;
    w1s[j * 129 + c] = W1[i];
  }
  if (tid < 64) b1s[tid] = b1[tid];
  if (tid < 128) w2s[tid] = W2[tid];
  if (tid < 2) b2s[tid] = b2[tid];

  const int wid = tid >> 5, lane = tid & 31;
  const int node = blockIdx.x * 8 + wid;
  float4 v = make_float4(0.f, 0.f, 0.f, 0.f);
  if (node < n) v = *(const float4*)(h + (size_t)node * 128 + lane * 4);
  *(float4*)(&rowbuf[wid][lane * 4]) = v;
  __syncthreads();

  const int j0 = lane, j1 = lane + 32;
  float a0 = 0.f, a1 = 0.f;
  for (int c = 0; c < 128; ++c) {
    float x = rowbuf[wid][c];
    a0 += x * w1s[j0 * 129 + c];
    a1 += x * w1s[j1 * 129 + c];
  }
  hidbuf[wid][j0] = fmaxf(a0 + b1s[j0], 0.f);
  hidbuf[wid][j1] = fmaxf(a1 + b1s[j1], 0.f);
  __syncthreads();

  if (lane < 2 && node < n) {
    float s = b2s[lane];
    for (int c = 0; c < 64; ++c) s += hidbuf[wid][c] * w2s[lane * 64 + c];
    outp[(size_t)node * 2 + lane] = s;
  }
}

// ---------------------------------------------------------------- launcher

extern "C" void kernel_launch(void* const* d_in, const int* in_sizes, int n_in,
                              void* d_out, int out_size, void* d_ws, size_t ws_size,
                              hipStream_t stream) {
  (void)in_sizes; (void)n_in; (void)out_size; (void)ws_size;

  const float* x  = (const float*)d_in[0];
  const int*   ei = (const int*)d_in[1];
  const float* Wn[3] = {(const float*)d_in[2], (const float*)d_in[7],  (const float*)d_in[12]};
  const float* Wr[3] = {(const float*)d_in[4], (const float*)d_in[9],  (const float*)d_in[14]};
  const float* g[3]  = {(const float*)d_in[5], (const float*)d_in[10], (const float*)d_in[15]};
  const float* be[3] = {(const float*)d_in[6], (const float*)d_in[11], (const float*)d_in[16]};
  const float* Wh1 = (const float*)d_in[17];
  const float* bh1 = (const float*)d_in[18];
  const float* Wh2 = (const float*)d_in[19];
  const float* bh2 = (const float*)d_in[20];

  char* ws = (char*)d_ws;
  size_t off = 0;
  auto alloc = [&](size_t bytes) {
    size_t o = off;
    off = (off + bytes + 255) & ~(size_t)255;
    return o;
  };
  _Float16* wsW = (_Float16*)(ws + alloc((size_t)3 * 2 * 64 * 32 * 16 * 2));  // 384 KB
  float* inv   = (float*)(ws + alloc((size_t)NNODES * 4));
  float* agg   = (float*)(ws + alloc((size_t)NNODES * 128 * 4));
  float* hbuf  = (float*)(ws + alloc((size_t)NNODES * 128 * 4));
  float* stats = (float*)(ws + alloc(256 * 4));
  float* ss    = (float*)(ws + alloc(256 * 4));

  // weight split (tiny)
  prep_w_kernel<<<(3 * 64 * 32 + 255) / 256, 256, 0, stream>>>(Wn[0], Wr[0], Wn[1], Wr[1], Wn[2], Wr[2], wsW);

  // degrees (edges identical for all layers)
  zero_kernel<<<1024, 256, 0, stream>>>(inv, NNODES);
  deg_kernel<<<(NEDGES + 255) / 256, 256, 0, stream>>>(ei, inv, NEDGES);
  inv_kernel<<<(NNODES + 255) / 256, 256, 0, stream>>>(inv, NNODES);

  const long tot = (long)NNODES * 128;
  for (int l = 0; l < 3; ++l) {
    const float* hin = (l == 0) ? x : hbuf;
    zero_kernel<<<2048, 256, 0, stream>>>(agg, tot);
    zero_kernel<<<1, 256, 0, stream>>>(stats, 256);
    scatter_kernel<<<(int)(((long)NEDGES * 32 + 255) / 256), 256, 0, stream>>>(hin, ei, agg, NEDGES);
    gemm_kernel<<<(NNODES + 63) / 64, 128, 0, stream>>>(agg, hin, inv, wsW + (size_t)l * 65536, agg, stats, NNODES);
    bn_finalize_kernel<<<1, 128, 0, stream>>>(stats, g[l], be[l], ss, 1.0f / (float)NNODES);
    bn_relu_kernel<<<(int)((tot + 255) / 256), 256, 0, stream>>>(agg, ss, hbuf, tot);
  }

  head_kernel<<<(NNODES + 7) / 8, 256, 0, stream>>>(hbuf, Wh1, bh1, Wh2, bh2, (float*)d_out, NNODES);
}
